// PointerDecoder_55087250539006
// MI455X (gfx1250) — matrix operand order repaired and checked by reading, not compiled, not run on hardware
//
#include <hip/hip_runtime.h>
#include <hip/hip_bf16.h>
#include <math.h>

// Problem constants (match the reference)
#define Bn 256
#define Ln 128     // == T
#define En 256
#define Hn 512
#define G4H (4 * Hn)   // 2048

typedef __attribute__((ext_vector_type(16))) __bf16 v16bf;
typedef __attribute__((ext_vector_type(8)))  float  v8f;

__device__ __forceinline__ unsigned short f2bf(float f) {
    __bf16 h = (__bf16)f;
    return __builtin_bit_cast(unsigned short, h);
}
__device__ __forceinline__ float bf2f(unsigned short u) {
    unsigned int v = ((unsigned int)u) << 16;
    return __builtin_bit_cast(float, v);
}

// ---------------------------------------------------------------------------
// Pack a row-major f32 weight W[N,K] (torch Linear layout) into a contiguous-k bf16
// B-fragment layout:  frag(kt,nt): lane l holds column n = nt*16 + (l&15),
// rows k = kt*32 + j + ((l&16)?16:0) for j = 0..15 (contiguous per lane-half).
// dst element index: (((kt*ntiles_stride + nt)*32 + l)*16 + j)
// ---------------------------------------------------------------------------
__global__ void pack_b_kernel(const float* __restrict__ W, int Ksrc,
                              unsigned short* __restrict__ dst,
                              int ktiles, int ntiles, int ntiles_stride) {
    int idx = blockIdx.x * blockDim.x + threadIdx.x;
    int total = ktiles * ntiles * 512;
    if (idx >= total) return;
    int j  = idx & 15;
    int l  = (idx >> 4) & 31;
    int nt = (idx >> 9) % ntiles;
    int kt = idx / (ntiles << 9);
    int n = nt * 16 + (l & 15);
    int k = kt * 32 + j + ((l & 16) ? 16 : 0);
    dst[(((size_t)kt * ntiles_stride + nt) * 32 + l) * 16 + j] =
        f2bf(W[(size_t)n * Ksrc + k]);
}

// ---------------------------------------------------------------------------
// WMMA GEMM with NB-way n-tile register blocking:
//   each wave computes a 16 x (16*NB) output tile; one A fragment per k-slice
//   is shared across NB WMMAs (cuts A re-reads and f32->bf16 cvt work NB-x).
// A is row-major f32, converted on the fly into a contiguous-k A layout matching packed B:
//   lanes 0-15 hold k_local {0..15}; lanes 16-31 hold {16..31};
//   element pairs (2*j2, 2*j2+1) are contiguous k, so float2 loads work.
// B comes pre-packed (see pack_b_kernel).
// ---------------------------------------------------------------------------
template <int NB>
__device__ __forceinline__ void mma_part(v8f* acc,
                                         const float* __restrict__ A, int lda,
                                         int mrow_base, int ktiles, int ktg_base,
                                         const unsigned short* __restrict__ Bpk,
                                         int ntiles_total, int nt0, int lane) {
    int m = lane & 15;
    const float* arow = A + (size_t)(mrow_base + m) * lda + ((lane & 16) ? 16 : 0);
    for (int kt = 0; kt < ktiles; ++kt) {
        const float* ar = arow + kt * 32;
        v16bf a;
#pragma unroll
        for (int j2 = 0; j2 < 8; ++j2) {
            int kf = 2 * j2;
            float2 f = *(const float2*)(ar + kf);
            a[2 * j2]     = (__bf16)f.x;
            a[2 * j2 + 1] = (__bf16)f.y;
        }
        const unsigned short* brow = Bpk +
            (((size_t)(ktg_base + kt) * ntiles_total + nt0) * 32 + lane) * 16;
        if (kt + 1 < ktiles)
            __builtin_prefetch((const void*)(brow + (size_t)ntiles_total * 512), 0, 3);
#pragma unroll
        for (int u = 0; u < NB; ++u) {
            const v16bf* bp = (const v16bf*)(brow + (size_t)u * 512);
            acc[u] = __builtin_amdgcn_wmma_f32_16x16x32_bf16(
                false, a, false, *bp, (short)0, acc[u], false, false);
        }
    }
}

template <int NB>
__global__ void wmma_gemm_kernel(const float* __restrict__ A1, int lda1, int kt1,
                                 const float* __restrict__ A2, int lda2, int kt2,
                                 const unsigned short* __restrict__ Bpk, int ntiles_total,
                                 const float* __restrict__ bias1,
                                 const float* __restrict__ bias2,
                                 float* __restrict__ Cf,
                                 unsigned short* __restrict__ Cbf,
                                 int mtiles, int ntiles, int N) {
    int ngroups = ntiles / NB;
    int group = blockIdx.x * (blockDim.x >> 5) + (threadIdx.x >> 5);
    if (group >= mtiles * ngroups) return;     // wave-uniform: EXEC stays full
    int mt  = group / ngroups;
    int nt0 = (group % ngroups) * NB;
    int lane = threadIdx.x & 31;

    v8f acc[NB];
#pragma unroll
    for (int u = 0; u < NB; ++u) acc[u] = (v8f){};

    mma_part<NB>(acc, A1, lda1, mt * 16, kt1, 0, Bpk, ntiles_total, nt0, lane);
    if (A2) mma_part<NB>(acc, A2, lda2, mt * 16, kt2, kt1, Bpk, ntiles_total, nt0, lane);

    int mbase = mt * 16 + ((lane & 16) ? 8 : 0);
#pragma unroll
    for (int u = 0; u < NB; ++u) {
        int ncol = (nt0 + u) * 16 + (lane & 15);
        float bv = 0.f;
        if (bias1) bv += bias1[ncol];
        if (bias2) bv += bias2[ncol];
#pragma unroll
        for (int r = 0; r < 8; ++r) {
            float v = acc[u][r] + bv;
            size_t off = (size_t)(mbase + r) * N + ncol;
            if (Cf) Cf[off] = v;
            else    Cbf[off] = f2bf(v);
        }
    }
}

// ---------------------------------------------------------------------------
// LSTM cell elementwise: gates [B,4H] order (i,f,g,o) -> update c,h in place
// ---------------------------------------------------------------------------
__global__ void lstm_cell_kernel(const float* __restrict__ gates,
                                 float* __restrict__ c, float* __restrict__ h) {
    int i = blockIdx.x * blockDim.x + threadIdx.x;
    if (i >= Bn * Hn) return;
    int b = i / Hn, hh = i % Hn;
    const float* g = gates + (size_t)b * G4H;
    float ig = g[hh], fg = g[Hn + hh], gg = g[2 * Hn + hh], og = g[3 * Hn + hh];
    float si = 1.f / (1.f + expf(-ig));
    float sf = 1.f / (1.f + expf(-fg));
    float so = 1.f / (1.f + expf(-og));
    float ct = sf * c[i] + si * tanhf(gg);
    c[i] = ct;
    h[i] = so * tanhf(ct);
}

// ---------------------------------------------------------------------------
// Attention step: energy = V . tanh(inp + ctx), mask, softmax, log_softmax,
// masked argmax (first-index tie-break), mask update, gather next x.
// One block (128 threads = 4 waves) per batch element.
// ---------------------------------------------------------------------------
__global__ void attention_kernel(const float* __restrict__ inp,            // [B,H]
                                 const unsigned short* __restrict__ ctxp,  // [B*L,H] bf16
                                 const float* __restrict__ V,              // [H]
                                 const float* __restrict__ emb,            // [B,L,E]
                                 float* __restrict__ mask,                 // [B,L]
                                 float* __restrict__ xbuf,                 // [B,E]
                                 float* __restrict__ outLA,                // outputs [B,T,L]
                                 float* __restrict__ ptrs,                 // pointers [B,T]
                                 int t) {
    int b = blockIdx.x;
    __shared__ float s_inp[Hn];
    __shared__ float s_V[Hn];
    __shared__ float s_e[Ln];
    __shared__ float s_red[Ln];
    __shared__ int   s_ridx[Ln];
    __shared__ int   s_idx;

    for (int i = threadIdx.x; i < Hn; i += blockDim.x) {
        s_inp[i] = inp[(size_t)b * Hn + i];
        s_V[i]   = V[i];
    }
    __syncthreads();

    int wave = threadIdx.x >> 5, lane = threadIdx.x & 31;
    for (int l = wave; l < Ln; l += 4) {
        const unsigned short* cp = ctxp + ((size_t)b * Ln + l) * Hn;
        float sum = 0.f;
        for (int hh = lane; hh < Hn; hh += 32)
            sum += s_V[hh] * tanhf(s_inp[hh] + bf2f(cp[hh]));
#pragma unroll
        for (int off = 16; off; off >>= 1) sum += __shfl_xor(sum, off, 32);
        if (lane == 0)
            s_e[l] = (mask[(size_t)b * Ln + l] == 0.f) ? -10.f : sum;
    }
    __syncthreads();

    int i = threadIdx.x;            // blockDim.x == Ln == 128
    // softmax(energy)
    s_red[i] = s_e[i]; __syncthreads();
    for (int st = 64; st > 0; st >>= 1) {
        if (i < st) s_red[i] = fmaxf(s_red[i], s_red[i + st]);
        __syncthreads();
    }
    float m1 = s_red[0]; __syncthreads();
    float ex = expf(s_e[i] - m1);
    s_red[i] = ex; __syncthreads();
    for (int st = 64; st > 0; st >>= 1) {
        if (i < st) s_red[i] += s_red[i + st];
        __syncthreads();
    }
    float alpha = ex / s_red[0]; __syncthreads();
    // log_softmax(alpha)  (faithful to the reference's double normalization)
    s_red[i] = alpha; __syncthreads();
    for (int st = 64; st > 0; st >>= 1) {
        if (i < st) s_red[i] = fmaxf(s_red[i], s_red[i + st]);
        __syncthreads();
    }
    float m2 = s_red[0]; __syncthreads();
    float ex2 = expf(alpha - m2);
    s_red[i] = ex2; __syncthreads();
    for (int st = 64; st > 0; st >>= 1) {
        if (i < st) s_red[i] += s_red[i + st];
        __syncthreads();
    }
    float la = alpha - m2 - logf(s_red[0]); __syncthreads();
    outLA[((size_t)b * Ln + t) * Ln + i] = la;
    // masked argmax, first index wins ties
    float mk = mask[(size_t)b * Ln + i];
    s_red[i]  = (mk == 0.f) ? -1e9f : la;
    s_ridx[i] = i;
    __syncthreads();
    for (int st = 64; st > 0; st >>= 1) {
        if (i < st) {
            float ov = s_red[i + st]; int oi = s_ridx[i + st];
            if (ov > s_red[i] || (ov == s_red[i] && oi < s_ridx[i])) {
                s_red[i] = ov; s_ridx[i] = oi;
            }
        }
        __syncthreads();
    }
    if (i == 0) {
        int idx = s_ridx[0];
        s_idx = idx;
        ptrs[(size_t)b * Ln + t] = (float)idx;
        mask[(size_t)b * Ln + idx] = 0.f;
    }
    __syncthreads();
    int idx = s_idx;
    for (int e = i; e < En; e += blockDim.x)
        xbuf[(size_t)b * En + e] = emb[((size_t)b * Ln + idx) * En + e];
}

// ---------------------------------------------------------------------------
// State init (every call: workspace must be rebuilt deterministically)
// ---------------------------------------------------------------------------
__global__ void init_state_kernel(const float* __restrict__ x0,
                                  const float* __restrict__ h0,
                                  const float* __restrict__ c0,
                                  float* __restrict__ xbuf, float* __restrict__ hbuf,
                                  float* __restrict__ cbuf, float* __restrict__ mask) {
    int i = blockIdx.x * blockDim.x + threadIdx.x;
    if (i < Bn * En) xbuf[i] = x0[i];
    if (i < Bn * Hn) { hbuf[i] = h0[i]; cbuf[i] = c0[i]; }
    if (i < Bn * Ln) mask[i] = 1.f;
}

__global__ void finalize_kernel(const float* __restrict__ h,
                                const float* __restrict__ c,
                                float* __restrict__ outH, float* __restrict__ outC) {
    int i = blockIdx.x * blockDim.x + threadIdx.x;
    if (i >= Bn * Hn) return;
    outH[i] = h[i];
    outC[i] = c[i];
}

// ---------------------------------------------------------------------------
extern "C" void kernel_launch(void* const* d_in, const int* in_sizes, int n_in,
                              void* d_out, int out_size, void* d_ws, size_t ws_size,
                              hipStream_t stream) {
    const float* emb   = (const float*)d_in[0];   // [B,L,E]
    const float* x0    = (const float*)d_in[1];   // [B,E]
    const float* h0    = (const float*)d_in[2];   // [B,H]
    const float* c0    = (const float*)d_in[3];   // [B,H]
    const float* ctx   = (const float*)d_in[4];   // [B,L,H]
    const float* W_ih  = (const float*)d_in[5];   // [4H,E]
    const float* b_ih  = (const float*)d_in[6];
    const float* W_hh  = (const float*)d_in[7];   // [4H,H]
    const float* b_hh  = (const float*)d_in[8];
    const float* W_in  = (const float*)d_in[9];   // [H,H]
    const float* b_in  = (const float*)d_in[10];
    const float* W_ctx = (const float*)d_in[11];  // [H,H]
    const float* b_ctx = (const float*)d_in[12];
    const float* V     = (const float*)d_in[13];  // [H]

    float* out = (float*)d_out;
    float* outLA = out;                                   // [B,T,L]
    float* outPtr = out + (size_t)Bn * Ln * Ln;           // [B,T]
    float* outH = outPtr + (size_t)Bn * Ln;               // [B,H]
    float* outC = outH + (size_t)Bn * Hn;                 // [B,H]

    // workspace carve (256B aligned)
    char* p = (char*)d_ws;
    auto carve = [&](size_t bytes) {
        void* r = (void*)p;
        p += (bytes + 255) & ~(size_t)255;
        return r;
    };
    float* xbuf  = (float*)carve((size_t)Bn * En * 4);
    float* hbuf  = (float*)carve((size_t)Bn * Hn * 4);
    float* cbuf  = (float*)carve((size_t)Bn * Hn * 4);
    float* gates = (float*)carve((size_t)Bn * G4H * 4);
    float* inp   = (float*)carve((size_t)Bn * Hn * 4);
    float* mask  = (float*)carve((size_t)Bn * Ln * 4);
    unsigned short* Bpk_ihhh = (unsigned short*)carve((size_t)24 * 128 * 512 * 2);
    unsigned short* Bpk_in   = (unsigned short*)carve((size_t)16 * 32 * 512 * 2);
    unsigned short* Bpk_ctx  = (unsigned short*)carve((size_t)16 * 32 * 512 * 2);
    unsigned short* ctxp     = (unsigned short*)carve((size_t)Bn * Ln * Hn * 2);

    // ---- pack weights (bf16 B-fragments) ----
    // W_ih -> ktiles 0..7 of combined, W_hh -> ktiles 8..23
    pack_b_kernel<<<(8 * 128 * 512) / 256, 256, 0, stream>>>(W_ih, En, Bpk_ihhh, 8, 128, 128);
    pack_b_kernel<<<(16 * 128 * 512) / 256, 256, 0, stream>>>(
        W_hh, Hn, Bpk_ihhh + (size_t)8 * 128 * 512, 16, 128, 128);
    pack_b_kernel<<<(16 * 32 * 512) / 256, 256, 0, stream>>>(W_in, Hn, Bpk_in, 16, 32, 32);
    pack_b_kernel<<<(16 * 32 * 512) / 256, 256, 0, stream>>>(W_ctx, Hn, Bpk_ctx, 16, 32, 32);

    // ---- init state ----
    init_state_kernel<<<(Bn * Hn) / 256, 256, 0, stream>>>(x0, h0, c0, xbuf, hbuf, cbuf, mask);

    // ---- context projection: ctxp[B*L,H] = ctx @ W_ctx^T + b_ctx (bf16 out) ----
    {
        int mtiles = (Bn * Ln) / 16, ntiles = Hn / 16;       // 2048 x 32
        int groups = mtiles * (ntiles / 4);                  // NB=4 -> 16384 waves
        wmma_gemm_kernel<4><<<(groups + 7) / 8, 256, 0, stream>>>(
            ctx, Hn, Hn / 32, nullptr, 0, 0, Bpk_ctx, 32,
            b_ctx, nullptr, nullptr, ctxp, mtiles, ntiles, Hn);
    }

    // ---- decode loop ----
    for (int t = 0; t < Ln; ++t) {
        // gates = x @ W_ih^T + h @ W_hh^T + b_ih + b_hh
        {
            int mtiles = Bn / 16, ntiles = G4H / 16;          // 16 x 128
            int groups = mtiles * (ntiles / 4);               // NB=4 -> 512 waves
            wmma_gemm_kernel<4><<<(groups + 7) / 8, 256, 0, stream>>>(
                xbuf, En, En / 32, hbuf, Hn, Hn / 32, Bpk_ihhh, 128,
                b_ih, b_hh, gates, nullptr, mtiles, ntiles, G4H);
        }
        lstm_cell_kernel<<<(Bn * Hn) / 256, 256, 0, stream>>>(gates, cbuf, hbuf);
        // inp = h @ W_in^T + b_in   (small: keep NB=1 for max wave parallelism)
        {
            int mtiles = Bn / 16, ntiles = Hn / 16;            // 16 x 32
            int groups = mtiles * ntiles;                      // 512 waves
            wmma_gemm_kernel<1><<<(groups + 7) / 8, 256, 0, stream>>>(
                hbuf, Hn, Hn / 32, nullptr, 0, 0, Bpk_in, 32,
                b_in, nullptr, inp, nullptr, mtiles, ntiles, Hn);
        }
        attention_kernel<<<Bn, Ln, 0, stream>>>(inp, ctxp, V, emb, mask,
                                                xbuf, outLA, outPtr, t);
    }

    finalize_kernel<<<(Bn * Hn) / 256, 256, 0, stream>>>(hbuf, cbuf, outH, outC);
}
